// Rwkv6SelfAttention_48644799595108
// MI455X (gfx1250) — matrix operand order repaired and checked by reading, not compiled, not run on hardware
//
#include <hip/hip_runtime.h>
#include <hip/hip_bf16.h>

// RWKV6 block for MI455X (gfx1250, wave32, WGP).
// - All GEMMs via v_wmma_f32_16x16x32_bf16 (bf16 in / f32 accumulate): the
//   projections are compute-bound (AI ~260 flop/B vs machine balance at
//   23.3 TB/s), and the bf16 WMMA carries 8x more K/instr than the f32 WMMA.
// - GEMM staging: branchless float4 loads + reg->LDS double buffering; the
//   compiler emits batched global_load_b128, one wait per tile, overlapped
//   with the WMMAs of the previous tile (verified in round-2 disasm).
// - Recurrent scan: split 4-ways along state columns (independent except the
//   wkv row-sum, reduced in the groupnorm kernel). Chunked k/r/td/v staging
//   now uses the CDNA5 async DMA path (global_load_async_to_lds_b128 +
//   s_wait_asynccnt) with LDS double buffering: chunk c+1's copies run behind
//   chunk c's compute without touching VGPRs or the VALU.

#define T_  1024
#define H_  2048
#define HS_ 64
#define NH_ 32
#define TM_ 32
#define TD_ 64

typedef __attribute__((ext_vector_type(16))) __bf16 v16bf;
typedef __attribute__((ext_vector_type(4)))  __bf16 v4bf;
typedef __attribute__((ext_vector_type(8)))  float  v8f;

union FragBF16 { uint4 u[2]; v16bf v; };

enum { ACT_NONE = 0, ACT_TANH = 1, ACT_SILU = 2, ACT_DECAY = 3, ACT_RESID = 4,
       ACT_X5 = 5 };

__device__ __forceinline__ float fast_tanh(float x) {
  float cx = fminf(fmaxf(x, -15.0f), 15.0f);
  float e  = __expf(2.0f * cx);
  return (e - 1.0f) / (e + 1.0f);
}

// CDNA5 async global->LDS copy (GV mode, 16B per lane, tracked by ASYNCcnt).
// LDS operand is the 32-bit LDS byte address (generic-pointer low bits).
__device__ __forceinline__ void async_copy_b128(const void* g, void* l) {
  unsigned loff = (unsigned)(size_t)l;
  asm volatile("global_load_async_to_lds_b128 %0, %1, off"
               :: "v"(loff), "v"(g) : "memory");
}
__device__ __forceinline__ void wait_async0() {
  asm volatile("s_wait_asynccnt 0x0" ::: "memory");
}

// ---------------------------------------------------------------------------
// bf16-WMMA GEMM: C[M,N] = act(A[M,K] @ B[K,N])
// Requires M%128==0, K%32==0, N%4==0 (true for every launch here).
// NGUARD: N may not be a multiple of 64 (only the N=160 t5 GEMM).
// Block 256 thr = 8 waves; tile 128x64, BK=32; wave tile 32x32 = 2x2 WMMA.
// ---------------------------------------------------------------------------
template <int ACT, bool NGUARD>
__global__ __launch_bounds__(256) void gemm_wmma(
    const float* __restrict__ A, int lda,
    const float* __restrict__ B, int ldb,
    float* __restrict__ C, int ldc,
    int M, int N, int K,
    const float* __restrict__ aux,    // DECAY bias / RESID residual / X5 maa
    const float* __restrict__ aux2,   // X5: xl
    const float* __restrict__ aux3)   // X5: sx
{
  // Pitch 40 bf16 (80 B): 16B-aligned uint4 fragment reads, 16 fragment rows
  // land on disjoint LDS bank quads.
  __shared__ __align__(16) __bf16 lA[2][128 * 40];
  __shared__ __align__(16) __bf16 lB[2][64 * 40];   // transposed: [n*40 + k]

  const int tid  = threadIdx.x;
  const int bM   = blockIdx.y * 128;
  const int bN   = blockIdx.x * 64;
  const int w    = tid >> 5;
  const int lane = tid & 31;
  const int wm   = w >> 1;
  const int wn   = w & 1;
  const int lr   = lane & 15;
  const int lh   = lane >> 4;

  float4 ar[4], br[2];

  auto loadTiles = [&](int k0) {
#pragma unroll
    for (int e = 0; e < 4; ++e) {               // A: 128x32 fp32 = 1024 float4
      int flat = tid + e * 256;
      int row = flat >> 3, c4 = (flat & 7) << 2;
      ar[e] = *(const float4*)(A + (size_t)(bM + row) * lda + k0 + c4);
    }
#pragma unroll
    for (int e = 0; e < 2; ++e) {               // B: 32x64 fp32 = 512 float4
      int flat = tid + e * 256;
      int kr = flat >> 4, n4 = (flat & 15) << 2;
      int gn = bN + n4;
      if (NGUARD) {
        bool ok = (gn + 4) <= N;                // N%4==0 -> all-or-nothing
        const float4* p =
            (const float4*)(B + (size_t)(k0 + kr) * ldb + (ok ? gn : 0));
        float4 v = *p;
        if (!ok) { v.x = 0.f; v.y = 0.f; v.z = 0.f; v.w = 0.f; }
        br[e] = v;
      } else {
        br[e] = *(const float4*)(B + (size_t)(k0 + kr) * ldb + gn);
      }
    }
  };
  auto storeTiles = [&](int buf) {
#pragma unroll
    for (int e = 0; e < 4; ++e) {
      int flat = tid + e * 256;
      int row = flat >> 3, c4 = (flat & 7) << 2;
      *(v4bf*)&lA[buf][row * 40 + c4] =
          (v4bf){(__bf16)ar[e].x, (__bf16)ar[e].y, (__bf16)ar[e].z, (__bf16)ar[e].w};
    }
#pragma unroll
    for (int e = 0; e < 2; ++e) {               // transpose into [n][k]
      int flat = tid + e * 256;
      int kr = flat >> 4, n4 = (flat & 15) << 2;
      lB[buf][(n4 + 0) * 40 + kr] = (__bf16)br[e].x;
      lB[buf][(n4 + 1) * 40 + kr] = (__bf16)br[e].y;
      lB[buf][(n4 + 2) * 40 + kr] = (__bf16)br[e].z;
      lB[buf][(n4 + 3) * 40 + kr] = (__bf16)br[e].w;
    }
  };

  v8f acc[2][2] = {};

  loadTiles(0);
  storeTiles(0);
  __syncthreads();

  int buf = 0;
  for (int k0 = 0; k0 < K; k0 += 32) {
    const bool more = (k0 + 32) < K;
    if (more) loadTiles(k0 + 32);               // overlap with WMMAs below

    FragBF16 fa[2], fb[2];
#pragma unroll
    for (int i = 0; i < 2; ++i) {
      // A: lane = row lane%16; K halves at lh*8 and lh*8+16.
      const __bf16* p = &lA[buf][(wm * 32 + i * 16 + lr) * 40 + lh * 8];
      fa[i].u[0] = *(const uint4*)(p);
      fa[i].u[1] = *(const uint4*)(p + 16);
    }
#pragma unroll
    for (int j = 0; j < 2; ++j) {
      // B: lane = col lane%16; K halves at lh*8 and lh*8+16.
      const __bf16* p = &lB[buf][(wn * 32 + j * 16 + lr) * 40 + lh * 8];
      fb[j].u[0] = *(const uint4*)(p);
      fb[j].u[1] = *(const uint4*)(p + 16);
    }
#pragma unroll
    for (int i = 0; i < 2; ++i)
#pragma unroll
      for (int j = 0; j < 2; ++j)
        acc[i][j] = __builtin_amdgcn_wmma_f32_16x16x32_bf16(
            false, fa[i].v, false, fb[j].v, (short)0, acc[i][j], false, false);

    if (more) {
      storeTiles(buf ^ 1);                      // other buffer: no race
      __syncthreads();
    }
    buf ^= 1;
  }

  // epilogue: VGPR e -> row = e + 8*lh, col = lane%16
#pragma unroll
  for (int i = 0; i < 2; ++i) {
#pragma unroll
    for (int j = 0; j < 2; ++j) {
      int col = bN + wn * 32 + j * 16 + lr;
      if (!NGUARD || col < N) {
#pragma unroll
        for (int e = 0; e < 8; ++e) {
          int row = bM + wm * 32 + i * 16 + lh * 8 + e;
          size_t idx = (size_t)row * ldc + col;
          float a = acc[i][j][e];
          if (ACT == ACT_TANH) {
            a = fast_tanh(a);
          } else if (ACT == ACT_SILU) {
            a = a / (1.0f + __expf(-a));
          } else if (ACT == ACT_DECAY) {
            a += aux[col];
            a = fminf(fmaxf(a, -9.72f), 2.27f);
            a = __expf(-__expf(a));
          } else if (ACT == ACT_RESID) {
            a += aux[idx];
          } else if (ACT == ACT_X5) {
            a = aux2[idx] + aux3[idx] * (a + aux[col]);
          }
          C[idx] = a;
        }
      }
    }
  }
}

// ---------------------------------------------------------------------------
// LayerNorm over H per token; emits state1_out = xl[T-1].
// ---------------------------------------------------------------------------
__global__ __launch_bounds__(256) void ln_kernel(
    const float* __restrict__ x, const float* __restrict__ w,
    const float* __restrict__ b, float* __restrict__ xl,
    float* __restrict__ state1_out)
{
  const int t   = blockIdx.x;
  const int tid = threadIdx.x;
  const float* row = x + (size_t)t * H_;

  float sum = 0.f, sq = 0.f;
  for (int c = tid; c < H_; c += 256) {
    float v = row[c];
    sum += v; sq += v * v;
  }
  __shared__ float s1[256], s2[256];
  s1[tid] = sum; s2[tid] = sq;
  __syncthreads();
  for (int m = 128; m > 0; m >>= 1) {
    if (tid < m) { s1[tid] += s1[tid + m]; s2[tid] += s2[tid + m]; }
    __syncthreads();
  }
  float mu   = s1[0] * (1.0f / H_);
  float var  = s2[0] * (1.0f / H_) - mu * mu;
  float rinv = rsqrtf(var + 1e-5f);

  float* o = xl + (size_t)t * H_;
  for (int c = tid; c < H_; c += 256) {
    float y = (row[c] - mu) * rinv * w[c] + b[c];
    o[c] = y;
    if (t == T_ - 1) state1_out[c] = y;
  }
}

// sx = past - xl (state1 for t==0, else xl[t-1]); xxx = xl + sx*time_maa_x
__global__ __launch_bounds__(256) void xxx_kernel(
    const float* __restrict__ xl, const float* __restrict__ state1,
    const float* __restrict__ tmx, float* __restrict__ sx,
    float* __restrict__ xxx)
{
  size_t idx = (size_t)blockIdx.x * 256 + threadIdx.x;
  int t = (int)(idx >> 11);
  int c = (int)(idx & (H_ - 1));
  float prev = (t == 0) ? state1[c] : xl[idx - H_];
  float cur  = xl[idx];
  float s    = prev - cur;
  sx[idx]  = s;
  xxx[idx] = cur + s * tmx[c];
}

// ---------------------------------------------------------------------------
// Recurrent scan, 4-way split along state columns.
// grid (NH, 4); 64 threads; thread = state row i, owns 16 columns in regs.
//   wkv_i += sum_{j in slab} (v_i*k_j*tf_j + s_ij)*r_j  (partial, per slab)
//   s_ij   = v_i*k_j + s_ij*td_j
// 32-step chunks of k/r/td/v staged by the async DMA path into
// double-buffered LDS; chunk c+1's copies overlap chunk c's compute.
// ---------------------------------------------------------------------------
#define LC_ 32   // steps per chunk
#define JS_ 16   // columns per slab

__global__ __launch_bounds__(64) void scan_kernel(
    const float* __restrict__ kk, const float* __restrict__ vv,
    const float* __restrict__ rr, const float* __restrict__ td,
    const float* __restrict__ tf, const float* __restrict__ s2_in,
    float* __restrict__ wkv_part,   // 4 planes of (T,H)
    float* __restrict__ s2_out)
{
  const int h     = blockIdx.x;
  const int jq    = blockIdx.y;
  const int i     = threadIdx.x;
  const int jbase = jq * JS_;

  float s[JS_], tfr[JS_];
  const float* s0 = s2_in + (size_t)h * HS_ * HS_ + (size_t)i * HS_ + jbase;
#pragma unroll
  for (int j = 0; j < JS_; ++j) s[j] = s0[j];
#pragma unroll
  for (int j = 0; j < JS_; ++j) tfr[j] = tf[h * HS_ + jbase + j];

  __shared__ __align__(16) float lk[2][LC_ * JS_], lrr[2][LC_ * JS_],
                                 ltd[2][LC_ * JS_], lv[2][LC_ * HS_];

  const size_t slabOff = (size_t)h * HS_ + jbase;
  const size_t vOff    = (size_t)h * HS_;

  // Issue one chunk's async copies (14 x b128 per lane = 14 KB per block).
  auto issueChunk = [&](int t0, int buf) {
#pragma unroll
    for (int e = 0; e < 2; ++e) {               // LC*JS/4 = 128 float4 slots
      int flat = i + e * 64;
      int row = flat >> 2, c4 = (flat & 3) << 2;
      size_t g = (size_t)(t0 + row) * H_ + slabOff + c4;
      async_copy_b128(kk + g, &lk[buf][row * JS_ + c4]);
      async_copy_b128(rr + g, &lrr[buf][row * JS_ + c4]);
      async_copy_b128(td + g, &ltd[buf][row * JS_ + c4]);
    }
#pragma unroll
    for (int e = 0; e < 8; ++e) {               // LC*HS/4 = 512 float4 slots
      int flat = i + e * 64;
      int row = flat >> 4, c4 = (flat & 15) << 2;
      async_copy_b128(vv + (size_t)(t0 + row) * H_ + vOff + c4,
                      &lv[buf][row * HS_ + c4]);
    }
  };

  issueChunk(0, 0);
  int buf = 0;
  for (int c = 0; c < T_ / LC_; ++c) {
    wait_async0();                   // own async copies landed in LDS
    __syncthreads();                 // all waves' copies visible; prior
                                     // buffer's readers are done
    if (c + 1 < T_ / LC_) issueChunk((c + 1) * LC_, buf ^ 1);

#pragma unroll 1
    for (int t16 = 0; t16 < LC_; ++t16) {
      float vi  = lv[buf][t16 * HS_ + i];
      float acc = 0.0f;
      const float* pk = &lk[buf][t16 * JS_];
      const float* pr = &lrr[buf][t16 * JS_];
      const float* pt = &ltd[buf][t16 * JS_];
#pragma unroll
      for (int j = 0; j < JS_; ++j) {
        float kj = pk[j], rj = pr[j], dj = pt[j];
        acc += (vi * kj * tfr[j] + s[j]) * rj;
        s[j] = vi * kj + s[j] * dj;
      }
      wkv_part[(size_t)jq * T_ * H_ + (size_t)(c * LC_ + t16) * H_ + h * HS_ + i] = acc;
    }
    buf ^= 1;
  }

  float* so = s2_out + (size_t)h * HS_ * HS_ + (size_t)i * HS_ + jbase;
#pragma unroll
  for (int j = 0; j < JS_; ++j) so[j] = s[j];
}

// GroupNorm per (t,h) over HS=64 (one wave32/head, 2 elems/lane) with 4-way
// partial-wkv reduction folded in, then gate by g.
__global__ __launch_bounds__(256) void gn_gate_kernel(
    const float* __restrict__ wkv_part, const float* __restrict__ g,
    const float* __restrict__ lnw, const float* __restrict__ lnb,
    float* __restrict__ out)
{
  const size_t P = (size_t)T_ * H_;
  const int t    = blockIdx.x;
  const int w    = threadIdx.x >> 5;
  const int lane = threadIdx.x & 31;
  const int h    = blockIdx.y * 8 + w;

  const float* p = wkv_part + (size_t)t * H_ + (size_t)h * HS_;
  float a0 = p[lane]      + p[P + lane]      + p[2 * P + lane]      + p[3 * P + lane];
  float a1 = p[lane + 32] + p[P + lane + 32] + p[2 * P + lane + 32] + p[3 * P + lane + 32];

  float sum = a0 + a1, sq = a0 * a0 + a1 * a1;
#pragma unroll
  for (int m = 16; m > 0; m >>= 1) {
    sum += __shfl_xor(sum, m, 32);
    sq  += __shfl_xor(sq,  m, 32);
  }
  float mu   = sum * (1.0f / HS_);
  float var  = sq * (1.0f / HS_) - mu * mu;
  float rinv = rsqrtf(var + 1e-5f);

  int c0 = h * HS_ + lane, c1 = c0 + 32;
  size_t idx = (size_t)t * H_ + c0;
  float y0 = (a0 - mu) * rinv * lnw[c0] + lnb[c0];
  float y1 = (a1 - mu) * rinv * lnw[c1] + lnb[c1];
  out[idx]      = y0 * g[idx];
  out[idx + 32] = y1 * g[idx + 32];
}

// ---------------------------------------------------------------------------
extern "C" void kernel_launch(void* const* d_in, const int* in_sizes, int n_in,
                              void* d_out, int out_size, void* d_ws, size_t ws_size,
                              hipStream_t stream)
{
  const float* x       = (const float*)d_in[0];
  const float* state1  = (const float*)d_in[1];
  const float* state2  = (const float*)d_in[2];
  const float* ln1_w   = (const float*)d_in[3];
  const float* ln1_b   = (const float*)d_in[4];
  const float* tmaa_x  = (const float*)d_in[5];
  const float* tmaa    = (const float*)d_in[6];
  const float* tmaa_w1 = (const float*)d_in[7];   // (H, 5*TM)
  const float* tmaa_w2 = (const float*)d_in[8];   // (5, TM, H)
  const float* tdec_w1 = (const float*)d_in[9];   // (H, TD)
  const float* tdec_w2 = (const float*)d_in[10];  // (TD, H)
  const float* tdec_p  = (const float*)d_in[11];  // (NH,1,HS) = flat H
  const float* tfirst  = (const float*)d_in[12];  // (NH,1,HS)
  const float* W_r     = (const float*)d_in[13];
  const float* W_k     = (const float*)d_in[14];
  const float* W_v     = (const float*)d_in[15];
  const float* W_g     = (const float*)d_in[16];
  const float* W_o     = (const float*)d_in[17];
  const float* lnx_w   = (const float*)d_in[18];
  const float* lnx_b   = (const float*)d_in[19];

  const size_t P = (size_t)T_ * H_;
  float* out_y  = (float*)d_out;
  float* out_s1 = out_y + P;
  float* out_s2 = out_s1 + H_;

  float* ws  = (float*)d_ws;
  float* xl  = ws;                                // P
  float* sx  = xl + P;                            // P
  float* xxx = sx + P;                            // P (reused as td)
  float* t5  = xxx + P;                           // T*5*TM
  float* x5  = t5 + (size_t)T_ * 5 * TM_;         // 5P (mix/x5, later wkv+gated)
  float* rb  = x5 + 5 * P;
  float* kb  = rb + P;
  float* vb  = kb + P;
  float* gb  = vb + P;
  float* td1 = gb + P;                            // T*TD
  float* td       = xxx;                          // reuse
  float* wkv_part = x5;                           // reuse planes 0..3
  float* gated    = x5 + 4 * P;                   // reuse plane 4

  // 1. LayerNorm + state1_out
  ln_kernel<<<T_, 256, 0, stream>>>(x, ln1_w, ln1_b, xl, out_s1);

  // 2. token shift
  xxx_kernel<<<(unsigned)(P / 256), 256, 0, stream>>>(xl, state1, tmaa_x, sx, xxx);

  // 3. t5 = tanh(xxx @ time_maa_w1)   M=1024 N=160 K=2048 (ragged N -> NGUARD)
  {
    dim3 grid((5 * TM_ + 63) / 64, T_ / 128);
    gemm_wmma<ACT_TANH, true><<<grid, 256, 0, stream>>>(
        xxx, H_, tmaa_w1, 5 * TM_, t5, 5 * TM_, T_, 5 * TM_, H_,
        nullptr, nullptr, nullptr);
  }

  // 4+5. x5[f] = xl + sx*(t5[:,f,:] @ w2[f] + time_maa[f])  (fused epilogue)
  {
    dim3 grid(H_ / 64, T_ / 128);
    for (int f = 0; f < 5; ++f)
      gemm_wmma<ACT_X5, false><<<grid, 256, 0, stream>>>(
          t5 + f * TM_, 5 * TM_, tmaa_w2 + (size_t)f * TM_ * H_, H_,
          x5 + (size_t)f * P, H_, T_, H_, TM_,
          tmaa + (size_t)f * H_, xl, sx);
  }

  // 6. projections   M=1024 N=2048 K=2048
  {
    dim3 grid(H_ / 64, T_ / 128);
    gemm_wmma<ACT_NONE, false><<<grid, 256, 0, stream>>>(
        x5 + 3 * P, H_, W_r, H_, rb, H_, T_, H_, H_, nullptr, nullptr, nullptr);
    gemm_wmma<ACT_NONE, false><<<grid, 256, 0, stream>>>(
        x5 + 1 * P, H_, W_k, H_, kb, H_, T_, H_, H_, nullptr, nullptr, nullptr);
    gemm_wmma<ACT_NONE, false><<<grid, 256, 0, stream>>>(
        x5 + 2 * P, H_, W_v, H_, vb, H_, T_, H_, H_, nullptr, nullptr, nullptr);
    gemm_wmma<ACT_SILU, false><<<grid, 256, 0, stream>>>(
        x5 + 4 * P, H_, W_g, H_, gb, H_, T_, H_, H_, nullptr, nullptr, nullptr);
  }

  // 7. td1 = tanh(x5[0] @ time_decay_w1)   N=64 K=2048
  {
    dim3 grid(TD_ / 64, T_ / 128);
    gemm_wmma<ACT_TANH, false><<<grid, 256, 0, stream>>>(
        x5, H_, tdec_w1, TD_, td1, TD_, T_, TD_, H_, nullptr, nullptr, nullptr);
  }

  // 8. td = exp(-exp(clip(td1 @ time_decay_w2 + time_decay_p)))   K=64
  {
    dim3 grid(H_ / 64, T_ / 128);
    gemm_wmma<ACT_DECAY, false><<<grid, 256, 0, stream>>>(
        td1, TD_, tdec_w2, H_, td, H_, T_, H_, TD_, tdec_p, nullptr, nullptr);
  }

  // 9. scan: 4 column slabs per head; partial wkv planes + state2_out
  {
    dim3 grid(NH_, 4);
    scan_kernel<<<grid, 64, 0, stream>>>(kb, vb, rb, td, tfirst, state2,
                                         wkv_part, out_s2);
  }

  // 10. groupnorm (+4-way partial sum) + gate
  {
    dim3 grid(T_, NH_ / 8);
    gn_gate_kernel<<<grid, 256, 0, stream>>>(wkv_part, gb, lnx_w, lnx_b, gated);
  }

  // 11. y = gated @ W_o + x (fused residual)
  {
    dim3 grid(H_ / 64, T_ / 128);
    gemm_wmma<ACT_RESID, false><<<grid, 256, 0, stream>>>(
        gated, H_, W_o, H_, out_y, H_, T_, H_, H_, x, nullptr, nullptr);
  }
}
